// SpectralCirculantLayer_86887188398194
// MI455X (gfx1250) — matrix operand order repaired and checked by reading, not compiled, not run on hardware
//
#include <hip/hip_runtime.h>

typedef __attribute__((ext_vector_type(8)))  float    v8f;
typedef __attribute__((ext_vector_type(8)))  _Float16 v8h;
typedef __attribute__((ext_vector_type(16))) _Float16 v16h;

#define N_FEAT  4096
#define BATCH   8192
#define KTRUNC  1024
#define TILE_M  128
#define TILE_N  128
#define TILE_K  32
#define LDSS    40   // padded LDS row stride in halves (80B: 16B-aligned, bank-spread)

// ---------------------------------------------------------------------------
// Kernel 1: build the real circulant filter h = ifft(W) from the truncated
// half-spectrum, store it time-REVERSED and duplicated as f16:
//   gd[i] = h[(-i) mod 4096],  i in [0, 8192)
// so that Bt[n][k] = h[(n-k) mod 4096] = gd[4096 + k - n]  (contiguous in k).
// ---------------------------------------------------------------------------
__global__ void build_filter_kernel(const float* __restrict__ wr,
                                    const float* __restrict__ wi,
                                    _Float16* __restrict__ gd) {
  int n = blockIdx.x * blockDim.x + threadIdx.x;
  if (n >= N_FEAT) return;
  float acc = wr[0];                       // DC bin (always real)
  const float w0 = 6.28318530717958647692f / (float)N_FEAT;
  for (int k = 1; k < KTRUNC; ++k) {       // k >= KTRUNC masked out (incl. Nyquist)
    int ph = (k * n) & (N_FEAT - 1);       // periodic phase reduction
    float s, c;
    __sincosf(w0 * (float)ph, &s, &c);
    acc += 2.0f * (wr[k] * c - wi[k] * s);
  }
  acc *= (1.0f / (float)N_FEAT);
  _Float16 hv = (_Float16)acc;
  int t = (N_FEAT - n) & (N_FEAT - 1);     // reversed index
  gd[t]          = hv;
  gd[t + N_FEAT] = hv;
}

// ---------------------------------------------------------------------------
// Kernel 2: y = x @ circulant(h) + bias via v_wmma_f32_16x16x32_f16.
// Workgroup tile 128x128, 8 waves (2x4), wave tile 64x32, K-step 32.
// Double-buffered LDS: stage tile i+1 while computing on tile i
// -> one workgroup barrier per k-step, global loads overlap WMMAs.
// ---------------------------------------------------------------------------
__global__ __launch_bounds__(256)
void circulant_gemm_kernel(const float* __restrict__ x,
                           const _Float16* __restrict__ gd,
                           const float* __restrict__ bias_p,
                           float* __restrict__ out) {
  __shared__ __align__(16) _Float16 As[2][TILE_M * LDSS];  // A tiles [m][k]
  __shared__ __align__(16) _Float16 Bs[2][TILE_N * LDSS];  // B tiles TRANSPOSED [n][k]

  const int tid  = threadIdx.x;
  const int lane = tid & 31;
  const int wave = tid >> 5;
  const int wm   = (wave >> 2) * 64;       // wave M offset inside WG tile
  const int wn   = (wave & 3)  * 32;       // wave N offset inside WG tile
  const int m0   = blockIdx.y * TILE_M;
  const int n0   = blockIdx.x * TILE_N;

  const int lhi = lane >> 4;               // 0: lanes 0-15, 1: lanes 16-31
  const int llo = lane & 15;

  const int srow  = tid >> 1;              // staging row 0..127
  const int shalf = (tid & 1) * 16;        // half-row: 0 or 16 halves

  // stage one 128x32 A tile (f32->f16) and one 128x32 Bt tile into buffer `buf`
  auto stage = [&](int k0, int buf) {
    // A: x[m0+srow][k0+shalf .. +16)  (coalesced 64B per thread)
    const float4* asrc =
        (const float4*)(x + (size_t)(m0 + srow) * N_FEAT + k0 + shalf);
    _Float16* adst = &As[buf][srow * LDSS + shalf];
#pragma unroll
    for (int v = 0; v < 4; ++v) {
      float4 f = asrc[v];
      adst[4 * v + 0] = (_Float16)f.x;
      adst[4 * v + 1] = (_Float16)f.y;
      adst[4 * v + 2] = (_Float16)f.z;
      adst[4 * v + 3] = (_Float16)f.w;
    }
    // Bt: Bs[n][k] = gd[4096 + (k0+k) - (n0+n)]  (16KB table, cache-hot)
    const _Float16* bsrc = gd + (N_FEAT + k0 + shalf - (n0 + srow));
    _Float16* bdst = &Bs[buf][srow * LDSS + shalf];
#pragma unroll
    for (int v = 0; v < 16; ++v) bdst[v] = bsrc[v];
  };

  v8f acc[4][2];
#pragma unroll
  for (int mf = 0; mf < 4; ++mf)
#pragma unroll
    for (int nf = 0; nf < 2; ++nf) acc[mf][nf] = (v8f){};

  stage(0, 0);                             // prologue: tile 0 into buffer 0

  for (int it = 0, k0 = 0; k0 < N_FEAT; k0 += TILE_K, ++it) {
    __syncthreads();                       // buffer `cur` fully staged
    const int cur = it & 1;

    if (k0 + TILE_K < N_FEAT)              // stage next tile into other buffer
      stage(k0 + TILE_K, cur ^ 1);         // (overlaps with WMMAs below)

    // ---- B fragments: lane<16 -> column llo K[0:16); lane>=16 -> K[16:32) ----
    v16h bfrag[2];
#pragma unroll
    for (int nf = 0; nf < 2; ++nf) {
      const _Float16* p = &Bs[cur][(wn + nf * 16 + llo) * LDSS + lhi * 16];
      v8h b0 = *(const v8h*)(p);
      v8h b1 = *(const v8h*)(p + 8);
#pragma unroll
      for (int i = 0; i < 8; ++i) { bfrag[nf][i] = b0[i]; bfrag[nf][i + 8] = b1[i]; }
    }

    // ---- A fragments + WMMA: lanes 0-15 K[0:16); lanes 16-31 K[16:32) ----
#pragma unroll
    for (int mf = 0; mf < 4; ++mf) {
      const _Float16* p = &As[cur][(wm + mf * 16 + llo) * LDSS + lhi * 16];
      v8h a0 = *(const v8h*)(p);
      v8h a1 = *(const v8h*)(p + 8);
      v16h afrag;
#pragma unroll
      for (int i = 0; i < 8; ++i) { afrag[i] = a0[i]; afrag[i + 8] = a1[i]; }
#pragma unroll
      for (int nf = 0; nf < 2; ++nf) {
        acc[mf][nf] = __builtin_amdgcn_wmma_f32_16x16x32_f16(
            /*neg_a=*/false, afrag, /*neg_b=*/false, bfrag[nf],
            /*c_mod=*/(short)0, acc[mf][nf],
            /*reuse_a=*/false, /*reuse_b=*/false);
      }
    }
  }

  // ---- epilogue: C/D layout -> lane<16: M=v, lane>=16: M=8+v; N = llo ----
  const float bias = *bias_p;
#pragma unroll
  for (int mf = 0; mf < 4; ++mf)
#pragma unroll
    for (int nf = 0; nf < 2; ++nf) {
      const int col  = n0 + wn + nf * 16 + llo;
      const int rowb = m0 + wm + mf * 16 + lhi * 8;
#pragma unroll
      for (int v = 0; v < 8; ++v)
        out[(size_t)(rowb + v) * N_FEAT + col] = acc[mf][nf][v] + bias;
    }
}

extern "C" void kernel_launch(void* const* d_in, const int* in_sizes, int n_in,
                              void* d_out, int out_size, void* d_ws, size_t ws_size,
                              hipStream_t stream) {
  const float* x    = (const float*)d_in[0];   // (8192, 4096) f32
  const float* wr   = (const float*)d_in[1];   // (2049,) f32
  const float* wi   = (const float*)d_in[2];   // (2049,) f32
  const float* bias = (const float*)d_in[3];   // scalar f32
  float*       out  = (float*)d_out;           // (8192, 4096) f32
  _Float16*    gd   = (_Float16*)d_ws;         // 8192 halves = 16 KB scratch

  // 1) spectrum -> reversed/duplicated f16 circulant filter
  build_filter_kernel<<<dim3(N_FEAT / 256), dim3(256), 0, stream>>>(wr, wi, gd);

  // 2) y = x @ circulant(h) + bias   (grid: N tiles in x, M tiles in y)
  circulant_gemm_kernel<<<dim3(N_FEAT / TILE_N, BATCH / TILE_M), dim3(256), 0, stream>>>(
      x, gd, bias, out);
}
